// MMoE_Layer_43250320671144
// MI455X (gfx1250) — matrix operand order repaired and checked by reading, not compiled, not run on hardware
//
#include <hip/hip_runtime.h>

#define B_ROWS   4096
#define D_IN     1024
#define H1       2048
#define H2       1024
#define NEXP     7
#define NSCN     7
#define XC_DIM   (D_IN + 16)

typedef __attribute__((ext_vector_type(16))) __bf16 v16bf;
typedef __attribute__((ext_vector_type(8)))  float  v8f;

union ABld { uint4 u[2]; v16bf v; };

__device__ __forceinline__ unsigned short f2bf(float f) {
    unsigned u = __float_as_uint(f);
    u += 0x7FFFu + ((u >> 16) & 1u);          // round-to-nearest-even
    return (unsigned short)(u >> 16);
}
__device__ __forceinline__ unsigned pack2(float a, float b) {
    return (unsigned)f2bf(a) | ((unsigned)f2bf(b) << 16);
}
__device__ __forceinline__ v8f v8zero() {
    v8f z;
#pragma unroll
    for (int i = 0; i < 8; ++i) z[i] = 0.0f;
    return z;
}

// --- CDNA5 async global->LDS copy (ASYNCcnt path), per CDNA5 ISA
__device__ __forceinline__ void async_ld_b128(unsigned lds_byte_off, const void* gptr) {
    asm volatile("global_load_async_to_lds_b128 %0, %1, off"
                 :: "v"(lds_byte_off), "v"(gptr) : "memory");
}
__device__ __forceinline__ void wait_async0() {
    __builtin_amdgcn_sched_barrier(0);        // pin preceding WMMAs before this stall
    asm volatile("s_wait_asynccnt 0x0" ::: "memory");
}
// split workgroup barrier WITHOUT the implicit loadcnt/storecnt fence of __syncthreads
__device__ __forceinline__ void wg_barrier() {
    asm volatile("s_barrier_signal -1\n\ts_barrier_wait -1" ::: "memory");
}

// ---------------------------------------------------------------------------
// Gating: one 64-thread block per batch row.
// ---------------------------------------------------------------------------
__global__ __launch_bounds__(64)
void gate_kernel(const float* __restrict__ x, const int* __restrict__ scene,
                 const float* __restrict__ S, const float* __restrict__ semb,
                 float* __restrict__ gate) {
    const int b = blockIdx.x;
    const int t = threadIdx.x;
    __shared__ float xc[XC_DIM];
    __shared__ float part[49 * 64];
    __shared__ float Gm[49];                 // [s*7 + e]
    __shared__ float v1s[NEXP], v2s[NEXP], ps[NEXP];

    const int sc = scene[b];
    for (int d = t; d < XC_DIM; d += 64)
        xc[d] = (d < D_IN) ? x[(size_t)b * D_IN + d] : semb[sc * 16 + (d - D_IN)];
    __syncthreads();

    float acc[49];
#pragma unroll
    for (int p = 0; p < 49; ++p) acc[p] = 0.0f;
    for (int d = t; d < XC_DIM; d += 64) {
        const float xv = xc[d];
        const float* Sd = S + (size_t)d * NEXP;
#pragma unroll
        for (int s = 0; s < NSCN; ++s)
#pragma unroll
            for (int e = 0; e < NEXP; ++e)
                acc[s * 7 + e] += xv * Sd[(size_t)s * XC_DIM * NEXP + e];
    }
#pragma unroll
    for (int p = 0; p < 49; ++p) part[p * 64 + t] = acc[p];
    __syncthreads();
    if (t < 49) {
        float s0 = 0.0f;
        for (int i = 0; i < 64; ++i) s0 += part[t * 64 + i];
        Gm[t] = s0;
    }
    __syncthreads();
    if (t < NEXP) {                           // per-expert softmax over 7 scenes
        const int e = t;
        float m = -1e30f;
        for (int s = 0; s < NSCN; ++s) m = fmaxf(m, Gm[s * 7 + e]);
        float se = 0.0f, msum = 0.0f;
        for (int s = 0; s < NSCN; ++s) { se += expf(Gm[s * 7 + e] - m); msum += Gm[s * 7 + e]; }
        const float lse = logf(se);
        ps[e]  = expf(Gm[sc * 7 + e] - m) / se;     // P[b,e,scene]
        v1s[e] = Gm[sc * 7 + e] - m - lse;          // logP at scene (pi ranking)
        v2s[e] = msum * (1.0f / 7.0f) - m - lse;    // mean_s logP  (qi ranking)
    }
    __syncthreads();
    if (t == 0) {
        int a1 = 0, a2 = 0;
        for (int e = 1; e < NEXP; ++e) {
            if (v1s[e] < v1s[a1]) a1 = e;
            if (v2s[e] < v2s[a2]) a2 = e;
        }
        float m = -1e30f;
        for (int e = 0; e < NEXP; ++e) m = fmaxf(m, ps[e]);
        float se = 0.0f;
        for (int e = 0; e < NEXP; ++e) se += expf(ps[e] - m);
        for (int e = 0; e < NEXP; ++e) {
            float g = expf(ps[e] - m) / se;
            if (e == a1 && e == a2) g = 0.0f;       // excluded only if worst in BOTH top-6 lists
            gate[(size_t)b * NEXP + e] = g;
        }
    }
}

// ---------------------------------------------------------------------------
// fp32 -> bf16 elementwise convert (x)
// ---------------------------------------------------------------------------
__global__ __launch_bounds__(256)
void convert_bf16(const float* __restrict__ in, unsigned short* __restrict__ out, long long n) {
    long long i = (long long)blockIdx.x * 256 + threadIdx.x;
    if (i < n) out[i] = f2bf(in[i]);
}

// ---------------------------------------------------------------------------
// Pre-tile weights into WMMA A-operand layout (A[M=outfeat][K=infeat] = W[K][M]).
// Tile storage: [e][mt = M/16][kt = K/32][lane 0..31][elem 0..15]
// lane l holds row M = l%16; elem i -> K = (i<8 ? i : i+8) + (l>=16 ? 8 : 0)
// ---------------------------------------------------------------------------
__global__ __launch_bounds__(256)
void tile_weights(const float* __restrict__ W, unsigned short* __restrict__ Wt,
                  int Ddim /*K*/, int Ndim /*M*/) {
    long long idx = (long long)blockIdx.x * 256 + threadIdx.x;
    const long long per_e = (long long)Ddim * Ndim;
    if (idx >= (long long)NEXP * per_e) return;
    const int e   = (int)(idx / per_e);
    long long rem = idx - (long long)e * per_e;
    const int d = (int)(rem / Ndim);          // K index
    const int j = (int)(rem % Ndim);          // M index
    const int kk = d & 31, kt = d >> 5;
    const int mt = j >> 4, l16 = j & 15;
    const int l = l16 + (((kk >> 3) & 1) ? 16 : 0);
    const int i = (kk & 7) + (kk >= 16 ? 8 : 0);
    const long long dst =
        ((((long long)e * (Ndim / 16) + mt) * (Ddim / 32) + kt) * 32 + l) * 16 + i;
    Wt[dst] = f2bf(W[idx]);
}

// ---------------------------------------------------------------------------
// GEMM1: h[e][b][j] = relu( x @ W1[e] + b1[e] ), bf16 out, row-major [e][B][H1].
// grid: (B/128, H1/128, 7); 256 threads = 8 waves (2 M-groups x 4 N-groups);
// wave tile 64(M) x 32(N); async double-buffered B staging, manual 2x unroll.
// ---------------------------------------------------------------------------
__global__ __launch_bounds__(256)
void gemm1_kernel(const unsigned short* __restrict__ W1t,
                  const unsigned short* __restrict__ xbf,
                  const float* __restrict__ b1,
                  unsigned short* __restrict__ hbf) {
    const int e  = blockIdx.z;
    const int n0 = blockIdx.x * 128;          // batch
    const int m0 = blockIdx.y * 128;          // H1 feature
    const int t = threadIdx.x, w = t >> 5, lane = t & 31;
    const int wm = w >> 2, wn = w & 3;
    const int l16 = lane & 15, lhi = lane >> 4;

    __shared__ unsigned short Bs0[128 * 40];  // double buffers (compile-time distinct)
    __shared__ unsigned short Bs1[128 * 40];
    const unsigned lds0 = (unsigned)(uintptr_t)(&Bs0[0]);
    const unsigned lds1 = (unsigned)(uintptr_t)(&Bs1[0]);

    const int r_a = t >> 2,         c_a = (t & 3) * 8;
    const int r_b = (t + 256) >> 2, c_b = ((t + 256) & 3) * 8;
    const unsigned o_a = (unsigned)(r_a * 40 + c_a) * 2u;
    const unsigned o_b = (unsigned)(r_b * 40 + c_b) * 2u;
    const unsigned short* xrow_a = xbf + (size_t)(n0 + r_a) * D_IN + c_a;
    const unsigned short* xrow_b = xbf + (size_t)(n0 + r_b) * D_IN + c_b;

    v8f acc[4][2];
#pragma unroll
    for (int ms = 0; ms < 4; ++ms)
#pragma unroll
        for (int ns = 0; ns < 2; ++ns) acc[ms][ns] = v8zero();

    const long long abase = ((long long)e * 128) * 32 * 512;

    auto computeB = [&](const unsigned short* Bbuf, ABld (&Av)[4]) {
        ABld Bv[2];
#pragma unroll
        for (int ns = 0; ns < 2; ++ns) {
            const unsigned short* p = Bbuf + (wn * 32 + ns * 16 + l16) * 40 + lhi * 8;
            Bv[ns].u[0] = *(const uint4*)p;
            Bv[ns].u[1] = *(const uint4*)(p + 16);
        }
#pragma unroll
        for (int ms = 0; ms < 4; ++ms)
#pragma unroll
            for (int ns = 0; ns < 2; ++ns)
                acc[ms][ns] = __builtin_amdgcn_wmma_f32_16x16x32_bf16(
                    false, Av[ms].v, false, Bv[ns].v, (short)0, acc[ms][ns], false, false);
    };

    // prologue: stage step 0 into buffer 0
    async_ld_b128(lds0 + o_a, xrow_a);
    async_ld_b128(lds0 + o_b, xrow_b);

    for (int i = 0; i < 32; i += 2) {
        // hoist A-tile loads for both halves (kt and kt+1 are +512 halfs apart)
        ABld Av0[4], Av1[4];
#pragma unroll
        for (int ms = 0; ms < 4; ++ms) {
            const int mt = (m0 >> 4) + wm * 4 + ms;
            const unsigned short* p0 =
                W1t + abase + (((long long)mt * 32 + i) << 9) + lane * 16;
            Av0[ms].u[0] = *(const uint4*)p0;
            Av0[ms].u[1] = *(const uint4*)(p0 + 8);
            Av1[ms].u[0] = *(const uint4*)(p0 + 512);
            Av1[ms].u[1] = *(const uint4*)(p0 + 520);
        }
        // half A: buffer 0, step i
        wait_async0();
        wg_barrier();
        async_ld_b128(lds1 + o_a, xrow_a + (i + 1) * 32);   // i+1 <= 31 always
        async_ld_b128(lds1 + o_b, xrow_b + (i + 1) * 32);
        computeB(Bs0, Av0);
        // half B: buffer 1, step i+1
        wait_async0();
        wg_barrier();
        if (i + 2 < 32) {
            async_ld_b128(lds0 + o_a, xrow_a + (i + 2) * 32);
            async_ld_b128(lds0 + o_b, xrow_b + (i + 2) * 32);
        }
        computeB(Bs1, Av1);
    }

    // epilogue: +bias, relu, -> bf16, store row-major h (this IS GEMM2's B layout)
#pragma unroll
    for (int ms = 0; ms < 4; ++ms) {
        const int f0 = m0 + wm * 64 + ms * 16 + lhi * 8;
        const float* bp = b1 + (size_t)e * H1 + f0;
        float bias[8];
#pragma unroll
        for (int r = 0; r < 8; ++r) bias[r] = bp[r];
#pragma unroll
        for (int ns = 0; ns < 2; ++ns) {
            const int brow = n0 + wn * 32 + ns * 16 + l16;
            float v[8];
#pragma unroll
            for (int r = 0; r < 8; ++r) {
                float f = acc[ms][ns][r] + bias[r];
                v[r] = f > 0.0f ? f : 0.0f;
            }
            uint4 st = make_uint4(pack2(v[0], v[1]), pack2(v[2], v[3]),
                                  pack2(v[4], v[5]), pack2(v[6], v[7]));
            *(uint4*)(hbf + ((size_t)e * B_ROWS + brow) * H1 + f0) = st;
        }
    }
}

// ---------------------------------------------------------------------------
// GEMM2: out[b][o] = sum_e gate[b,e] * relu( h[e] @ W2[e] + b2[e] ); dup to 2 tasks.
// Flattened (expert x K) 448-step async pipeline, manual 2x unroll
// (s is even => both halves are inside the same expert).
// ---------------------------------------------------------------------------
__global__ __launch_bounds__(256)
void gemm2_kernel(const unsigned short* __restrict__ W2t,
                  const unsigned short* __restrict__ hbf,
                  const float* __restrict__ b2,
                  const float* __restrict__ gate,
                  float* __restrict__ out) {
    const int n0 = blockIdx.x * 128;          // batch
    const int m0 = blockIdx.y * 128;          // H2 feature
    const int t = threadIdx.x, w = t >> 5, lane = t & 31;
    const int wm = w >> 2, wn = w & 3;
    const int l16 = lane & 15, lhi = lane >> 4;

    __shared__ unsigned short Bs0[128 * 40];
    __shared__ unsigned short Bs1[128 * 40];
    const unsigned lds0 = (unsigned)(uintptr_t)(&Bs0[0]);
    const unsigned lds1 = (unsigned)(uintptr_t)(&Bs1[0]);

    const int r_a = t >> 2,         c_a = (t & 3) * 8;
    const int r_b = (t + 256) >> 2, c_b = ((t + 256) & 3) * 8;
    const unsigned o_a = (unsigned)(r_a * 40 + c_a) * 2u;
    const unsigned o_b = (unsigned)(r_b * 40 + c_b) * 2u;

    v8f fin[4][2];
#pragma unroll
    for (int ms = 0; ms < 4; ++ms)
#pragma unroll
        for (int ns = 0; ns < 2; ++ns) fin[ms][ns] = v8zero();
    v8f acc[4][2];
#pragma unroll
    for (int ms = 0; ms < 4; ++ms)
#pragma unroll
        for (int ns = 0; ns < 2; ++ns) acc[ms][ns] = v8zero();

    const int TOT = NEXP * 64;                // 448 steps (7 experts x 64 K-steps)

    auto issue = [&](unsigned dstBase, int step) {
        const int ee = step >> 6;
        const int k1 = (step & 63) * 32;
        const unsigned short* hb = hbf + (size_t)ee * B_ROWS * H1;
        async_ld_b128(dstBase + o_a, hb + (size_t)(n0 + r_a) * H1 + k1 + c_a);
        async_ld_b128(dstBase + o_b, hb + (size_t)(n0 + r_b) * H1 + k1 + c_b);
    };
    auto computeB = [&](const unsigned short* Bbuf, ABld (&Av)[4]) {
        ABld Bv[2];
#pragma unroll
        for (int ns = 0; ns < 2; ++ns) {
            const unsigned short* p = Bbuf + (wn * 32 + ns * 16 + l16) * 40 + lhi * 8;
            Bv[ns].u[0] = *(const uint4*)p;
            Bv[ns].u[1] = *(const uint4*)(p + 16);
        }
#pragma unroll
        for (int ms = 0; ms < 4; ++ms)
#pragma unroll
            for (int ns = 0; ns < 2; ++ns)
                acc[ms][ns] = __builtin_amdgcn_wmma_f32_16x16x32_bf16(
                    false, Av[ms].v, false, Bv[ns].v, (short)0, acc[ms][ns], false, false);
    };

    // prologue: stage step 0 (expert 0, k0 = 0)
    issue(lds0, 0);

    for (int s = 0; s < TOT; s += 2) {
        const int e  = s >> 6;                // same expert for both halves (s even)
        const int kt = s & 63;
        const long long abase = ((long long)e * 64) * 64 * 512;

        ABld Av0[4], Av1[4];
#pragma unroll
        for (int ms = 0; ms < 4; ++ms) {
            const int mt = (m0 >> 4) + wm * 4 + ms;
            const unsigned short* p0 =
                W2t + abase + (((long long)mt * 64 + kt) << 9) + lane * 16;
            Av0[ms].u[0] = *(const uint4*)p0;
            Av0[ms].u[1] = *(const uint4*)(p0 + 8);
            Av1[ms].u[0] = *(const uint4*)(p0 + 512);
            Av1[ms].u[1] = *(const uint4*)(p0 + 520);
        }
        // half A: buffer 0, step s
        wait_async0();
        wg_barrier();
        issue(lds1, s + 1);                    // s+1 <= TOT-1 always
        computeB(Bs0, Av0);
        // half B: buffer 1, step s+1
        wait_async0();
        wg_barrier();
        if (s + 2 < TOT) issue(lds0, s + 2);
        computeB(Bs1, Av1);

        if (kt == 62) {                        // step s+1 was kt==63: expert done
            float gv[2];
#pragma unroll
            for (int ns = 0; ns < 2; ++ns)
                gv[ns] = gate[(size_t)(n0 + wn * 32 + ns * 16 + l16) * NEXP + e];
#pragma unroll
            for (int ms = 0; ms < 4; ++ms) {
                const int f0 = m0 + wm * 64 + ms * 16 + lhi * 8;
                const float* bp = b2 + (size_t)e * H2 + f0;
                float bias[8];
#pragma unroll
                for (int r = 0; r < 8; ++r) bias[r] = bp[r];
#pragma unroll
                for (int ns = 0; ns < 2; ++ns)
#pragma unroll
                    for (int r = 0; r < 8; ++r) {
                        float f = acc[ms][ns][r] + bias[r];
                        f = f > 0.0f ? f : 0.0f;
                        fin[ms][ns][r] += gv[ns] * f;
                        acc[ms][ns][r] = 0.0f;   // reset for next expert
                    }
            }
        }
    }

    // store both (identical) task outputs: out is [2][B][H2]
#pragma unroll
    for (int ms = 0; ms < 4; ++ms) {
        const int f0 = m0 + wm * 64 + ms * 16 + lhi * 8;
#pragma unroll
        for (int ns = 0; ns < 2; ++ns) {
            const int brow = n0 + wn * 32 + ns * 16 + l16;
            float* o0 = out + (size_t)brow * H2 + f0;
            float4 lo = make_float4(fin[ms][ns][0], fin[ms][ns][1], fin[ms][ns][2], fin[ms][ns][3]);
            float4 hi = make_float4(fin[ms][ns][4], fin[ms][ns][5], fin[ms][ns][6], fin[ms][ns][7]);
            *(float4*)o0 = lo;
            *(float4*)(o0 + 4) = hi;
            float* o1 = o0 + (size_t)B_ROWS * H2;
            *(float4*)o1 = lo;
            *(float4*)(o1 + 4) = hi;
        }
    }
}

// ---------------------------------------------------------------------------
extern "C" void kernel_launch(void* const* d_in, const int* in_sizes, int n_in,
                              void* d_out, int out_size, void* d_ws, size_t ws_size,
                              hipStream_t stream) {
    const float* x     = (const float*)d_in[0];
    const int*   scene = (const int*)  d_in[1];
    const float* W1    = (const float*)d_in[2];
    const float* b1    = (const float*)d_in[3];
    const float* W2    = (const float*)d_in[4];
    const float* b2    = (const float*)d_in[5];
    const float* S     = (const float*)d_in[6];
    const float* semb  = (const float*)d_in[7];
    float* out = (float*)d_out;

    auto alignup = [](size_t v) { return (v + 255) & ~(size_t)255; };
    size_t off = 0;
    unsigned short* xbf = (unsigned short*)((char*)d_ws + off);
    off += alignup((size_t)B_ROWS * D_IN * 2);
    unsigned short* w1t = (unsigned short*)((char*)d_ws + off);
    off += alignup((size_t)NEXP * D_IN * H1 * 2);
    unsigned short* w2t = (unsigned short*)((char*)d_ws + off);
    off += alignup((size_t)NEXP * H1 * H2 * 2);
    unsigned short* hbf = (unsigned short*)((char*)d_ws + off);
    off += alignup((size_t)NEXP * B_ROWS * H1 * 2);
    float* gate = (float*)((char*)d_ws + off);
    off += alignup((size_t)B_ROWS * NEXP * 4);
    (void)ws_size; (void)in_sizes; (void)n_in; (void)out_size;

    // gating (independent of GEMM path)
    gate_kernel<<<B_ROWS, 64, 0, stream>>>(x, scene, S, semb, gate);

    // bf16 conversions / weight tiling
    {
        long long n = (long long)B_ROWS * D_IN;
        convert_bf16<<<(unsigned)((n + 255) / 256), 256, 0, stream>>>(x, xbf, n);
    }
    {
        long long n = (long long)NEXP * D_IN * H1;
        tile_weights<<<(unsigned)((n + 255) / 256), 256, 0, stream>>>(W1, w1t, D_IN, H1);
    }
    {
        long long n = (long long)NEXP * H1 * H2;
        tile_weights<<<(unsigned)((n + 255) / 256), 256, 0, stream>>>(W2, w2t, H1, H2);
    }

    // expert layer 1: h = relu(x @ W1 + b1)  (bf16 out, WMMA)
    gemm1_kernel<<<dim3(B_ROWS / 128, H1 / 128, NEXP), 256, 0, stream>>>(w1t, xbf, b1, hbf);

    // expert layer 2 + gate-weighted mixture + dual-task store (WMMA)
    gemm2_kernel<<<dim3(B_ROWS / 128, H2 / 128, 1), 256, 0, stream>>>(w2t, hbf, b2, gate, out);
}
